// UniversalMatrixAutoencoder_44513041056231
// MI455X (gfx1250) — compile-verified, matrix operand order checked by reading, not run on hardware
//
#include <hip/hip_runtime.h>
#include <hip/hip_bf16.h>

// ---------------------------------------------------------------------------
// Fused universal-matrix autoencoder for MI455X (gfx1250, wave32, WMMA).
//   z     = relu(x@W1+b1)@W2 + b2            [B, 512]
//   recon = (sum over NF of z chunks) @ D^T  [B, 4096]
// All GEMMs via v_wmma_f32_16x16x32_bf16 (f32 accumulate). Weights are
// prepacked into WMMA B-fragment order (bf16) in d_ws so the hot loops do
// only contiguous 32B fragment loads from L2.
// ---------------------------------------------------------------------------

typedef __attribute__((ext_vector_type(16))) __bf16 bf16x16;
typedef __attribute__((ext_vector_type(8)))  float  f32x8;

#define WMMA_BF16(a, b, c) \
  __builtin_amdgcn_wmma_f32_16x16x32_bf16(false, (a), false, (b), (short)0, (c), false, false)

constexpr int B_  = 8192;   // batch
constexpr int D_  = 4096;   // input dim
constexpr int H_  = 128;    // hidden
constexpr int NZ_ = 512;    // NF*L
constexpr int L_  = 32;     // latent per factor
constexpr int KS1 = D_ / 32;   // 128 k-steps for GEMM1

// LDS row strides (padded: stride%64 != 0 -> no bank conflicts; 16B aligned)
constexpr int HSTR = 132;
constexpr int ZSTR = 36;

// packed fragment buffers in d_ws (element counts, bf16)
constexpr size_t W1P_ELEMS = (size_t)KS1 * 8  * 32 * 16;  // 524288  (1 MB)
constexpr size_t W2P_ELEMS = (size_t)4   * 32 * 32 * 16;  // 65536   (128 KB)
constexpr size_t DCP_ELEMS = (size_t)256 * 32 * 16;       // 131072  (256 KB)

// Fragment k-index within a 16x16x32 bf16 B/A fragment:
//   element e (0..15) of lane `lane` holds k_local = (e>>3)*16 + (lane>>4)*8 + (e&7)
__device__ __host__ __forceinline__ int frag_klocal(int lane, int e) {
  return ((e >> 3) << 4) + ((lane >> 4) << 3) + (e & 7);
}

// ---- prepack kernels: f32 weights -> bf16 fragment-ordered ----------------

__global__ void pack_w1(const float* __restrict__ W1, __bf16* __restrict__ out) {
  int tid  = blockIdx.x * 256 + threadIdx.x;        // 524288 total
  int e    = tid & 15;
  int lane = (tid >> 4) & 31;
  int ct   = (tid >> 9) & 7;
  int ks   = tid >> 12;
  int n    = ct * 16 + (lane & 15);
  int k    = ks * 32 + frag_klocal(lane, e);
  out[tid] = (__bf16)W1[(size_t)k * H_ + n];
}

__global__ void pack_w2(const float* __restrict__ W2, __bf16* __restrict__ out) {
  int tid  = blockIdx.x * 256 + threadIdx.x;        // 65536 total
  int e    = tid & 15;
  int lane = (tid >> 4) & 31;
  int ct   = (tid >> 9) & 31;
  int ks   = tid >> 14;
  int n    = ct * 16 + (lane & 15);
  int k    = ks * 32 + frag_klocal(lane, e);
  out[tid] = (__bf16)W2[(size_t)k * NZ_ + n];
}

__global__ void pack_dec(const float* __restrict__ dec, __bf16* __restrict__ out) {
  int tid  = blockIdx.x * 256 + threadIdx.x;        // 131072 total
  int e    = tid & 15;
  int lane = (tid >> 4) & 31;
  int ct   = tid >> 9;                              // 0..255 output col tile
  int d    = ct * 16 + (lane & 15);                 // output column (g*64 + k)
  int l    = frag_klocal(lane, e);                  // latent index 0..31
  out[tid] = (__bf16)dec[(size_t)d * L_ + l];       // decoders[g][k][l] flattened
}

// ---- fused main kernel: one wave handles 16 batch rows end-to-end ---------

__global__ __launch_bounds__(32) void fused_ae(
    const float*  __restrict__ x,
    const float*  __restrict__ b1,
    const float*  __restrict__ b2,
    const __bf16* __restrict__ w1p,
    const __bf16* __restrict__ w2p,
    const __bf16* __restrict__ dcp,
    float* __restrict__ zout,
    float* __restrict__ rout)
{
  __shared__ float sh_h[16 * HSTR];   // h tile [16 x 128], padded rows
  __shared__ float sh_z[16 * ZSTR];   // z_sum tile [16 x 32], padded rows

  const int lane = threadIdx.x;       // 0..31
  const int n    = lane & 15;
  const int hf   = lane >> 4;
  const int row0 = blockIdx.x * 16;   // 512 blocks cover B=8192

  const f32x8 zero8 = {};

  // ---------------- GEMM1: h = relu(x @ W1 + b1), [16 x 128] --------------
  f32x8 acc[8];
#pragma unroll
  for (int t = 0; t < 8; ++t) acc[t] = zero8;

  const float* xrow = x + (size_t)(row0 + n) * D_ + hf * 8;
#pragma unroll 1
  for (int ks = 0; ks < KS1; ++ks) {
    const float* p = xrow + ks * 32;
    __builtin_prefetch(p + 64, 0, 3);               // global_prefetch_b8, 2 steps ahead
    bf16x16 a;
#pragma unroll
    for (int j = 0; j < 8; ++j) {
      a[j]     = (__bf16)p[j];                      // k_local = hf*8 + j
      a[8 + j] = (__bf16)p[16 + j];                 // k_local = 16 + hf*8 + j
    }
    const __bf16* bp = w1p + (size_t)ks * 8 * 512 + lane * 16;
#pragma unroll
    for (int ct = 0; ct < 8; ++ct) {
      bf16x16 b = *(const bf16x16*)(bp + ct * 512);
      acc[ct] = WMMA_BF16(a, b, acc[ct]);
    }
  }

  // bias + ReLU, stash h tile in LDS (C-layout -> row-major)
#pragma unroll
  for (int ct = 0; ct < 8; ++ct) {
    float bv = b1[ct * 16 + n];
#pragma unroll
    for (int v = 0; v < 8; ++v) {
      float hv = acc[ct][v] + bv;
      hv = hv > 0.0f ? hv : 0.0f;
      sh_h[(v + 8 * hf) * HSTR + ct * 16 + n] = hv;
    }
  }
  __syncthreads();

  // ---------------- GEMM2: z = h @ W2 + b2, [16 x 512] --------------------
  f32x8 zs[2];                                      // z_sum tiles (latent 0-15 / 16-31)
  zs[0] = zero8; zs[1] = zero8;

#pragma unroll 1
  for (int nc = 0; nc < 4; ++nc) {                  // 128 output cols per chunk
    f32x8 zacc[8];
#pragma unroll
    for (int t = 0; t < 8; ++t) zacc[t] = zero8;

#pragma unroll
    for (int ks = 0; ks < 4; ++ks) {
      const float* hp = sh_h + n * HSTR + ks * 32 + hf * 8;
      bf16x16 a;
#pragma unroll
      for (int j = 0; j < 8; ++j) {
        a[j]     = (__bf16)hp[j];
        a[8 + j] = (__bf16)hp[16 + j];
      }
      const __bf16* bp = w2p + (size_t)(ks * 32 + nc * 8) * 512 + lane * 16;
#pragma unroll
      for (int ct = 0; ct < 8; ++ct) {
        bf16x16 b = *(const bf16x16*)(bp + ct * 512);
        zacc[ct] = WMMA_BF16(a, b, zacc[ct]);
      }
    }

    // epilogue: bias, emit z, fold factor-sum into zs (latent = col & 31)
#pragma unroll
    for (int ct = 0; ct < 8; ++ct) {
      int col = nc * 128 + ct * 16 + n;
      float bv = b2[col];
      const int t = ct & 1;
#pragma unroll
      for (int v = 0; v < 8; ++v) {
        float zv = zacc[ct][v] + bv;
        zout[(size_t)(row0 + v + 8 * hf) * NZ_ + col] = zv;
        zs[t][v] += zv;
      }
    }
  }

  // stash z_sum [16 x 32] in LDS, reload as A-fragment
#pragma unroll
  for (int t = 0; t < 2; ++t)
#pragma unroll
    for (int v = 0; v < 8; ++v)
      sh_z[(v + 8 * hf) * ZSTR + t * 16 + n] = zs[t][v];
  __syncthreads();

  bf16x16 az;
  {
    const float* zp = sh_z + n * ZSTR + hf * 8;
#pragma unroll
    for (int j = 0; j < 8; ++j) {
      az[j]     = (__bf16)zp[j];
      az[8 + j] = (__bf16)zp[16 + j];
    }
  }

  // ---------------- recon = z_sum @ D^T, [16 x 4096] ----------------------
  float* rbase = rout + (size_t)row0 * D_;
#pragma unroll 2
  for (int ct = 0; ct < 256; ++ct) {
    bf16x16 b = *(const bf16x16*)(dcp + (size_t)ct * 512 + lane * 16);
    f32x8 c = zero8;
    c = WMMA_BF16(az, b, c);
    float* rp = rbase + ct * 16 + n;
#pragma unroll
    for (int v = 0; v < 8; ++v)
      rp[(size_t)(v + 8 * hf) * D_] = c[v];
  }
}

// ---------------------------------------------------------------------------

extern "C" void kernel_launch(void* const* d_in, const int* in_sizes, int n_in,
                              void* d_out, int out_size, void* d_ws, size_t ws_size,
                              hipStream_t stream) {
  const float* x   = (const float*)d_in[0];
  const float* W1  = (const float*)d_in[1];
  const float* b1  = (const float*)d_in[2];
  const float* W2  = (const float*)d_in[3];
  const float* b2  = (const float*)d_in[4];
  const float* dec = (const float*)d_in[5];

  char* ws = (char*)d_ws;
  __bf16* w1p = (__bf16*)ws;
  __bf16* w2p = (__bf16*)(ws + W1P_ELEMS * 2);
  __bf16* dcp = (__bf16*)(ws + (W1P_ELEMS + W2P_ELEMS) * 2);

  float* zout = (float*)d_out;
  float* rout = zout + (size_t)B_ * NZ_;

  pack_w1<<<(int)(W1P_ELEMS / 256), 256, 0, stream>>>(W1, w1p);
  pack_w2<<<(int)(W2P_ELEMS / 256), 256, 0, stream>>>(W2, w2p);
  pack_dec<<<(int)(DCP_ELEMS / 256), 256, 0, stream>>>(dec, dcp);

  fused_ae<<<B_ / 16, 32, 0, stream>>>(x, b1, b2, w1p, w2p, dcp, zout, rout);
}
